// BiDAF2_11922829214354
// MI455X (gfx1250) — matrix operand order repaired and checked by reading, not compiled, not run on hardware
//
#include <hip/hip_runtime.h>
#include <hip/hip_bf16.h>
#include <stdint.h>

#define B_ 16
#define L_ 1024
#define D_ 768

typedef __bf16 bf16;
typedef __attribute__((ext_vector_type(8)))  bf16  v8bf;
typedef __attribute__((ext_vector_type(16))) bf16  v16bf;
typedef __attribute__((ext_vector_type(8)))  float v8f;

__device__ __forceinline__ unsigned short f2bf(float f) {
  union { float f; unsigned int u; } c; c.f = f;
  unsigned int u = c.u;
  u += 0x7FFFu + ((u >> 16) & 1u);   // round-to-nearest-even
  return (unsigned short)(u >> 16);
}

// A-operand (16x32, MxK, bf16): lane l holds row (l&15), K = b..b+7 and b+16..b+23, b=8*(l>>4).
// p points at row start + k + 8*(l>>4) halves; chunks at +0 and +16 halves.
__device__ __forceinline__ v16bf load_a_tile(const unsigned short* p) {
  v8bf lo = *reinterpret_cast<const v8bf*>(p);
  v8bf hi = *reinterpret_cast<const v8bf*>(p + 16);
  v16bf r;
#pragma unroll
  for (int i = 0; i < 8; ++i) { r[i] = lo[i]; r[i + 8] = hi[i]; }
  return r;
}
// B-operand for X @ Y^T: lane l holds column n=(l&15) = row n of Y, K = b..b+7 and b+16..b+23, b=8*(l>>4).
__device__ __forceinline__ v16bf load_b_tile(const unsigned short* p) {
  v8bf lo = *reinterpret_cast<const v8bf*>(p);
  v8bf hi = *reinterpret_cast<const v8bf*>(p + 16);
  v16bf r;
#pragma unroll
  for (int i = 0; i < 8; ++i) { r[i] = lo[i]; r[i + 8] = hi[i]; }
  return r;
}

// ---------------- Kernel 0: bf16 prep (h, q, qk = q + w1_w, w2_w) ----------------
__global__ void prep_kernel(const float* __restrict__ h, const float* __restrict__ q,
                            const float* __restrict__ w1w, const float* __restrict__ w2w,
                            unsigned short* __restrict__ hb, unsigned short* __restrict__ qb,
                            unsigned short* __restrict__ qkb, unsigned short* __restrict__ w2b) {
  size_t idx = (size_t)blockIdx.x * blockDim.x + threadIdx.x;
  if (idx >= (size_t)B_ * L_ * D_) return;
  unsigned int r = (unsigned int)(idx % ((size_t)L_ * D_));   // t*D + d
  float hv = h[idx], qv = q[idx];
  hb[idx]  = f2bf(hv);
  qb[idx]  = f2bf(qv);
  qkb[idx] = f2bf(qv + w1w[r]);
  if (idx < (size_t)L_ * D_) w2b[idx] = f2bf(w2w[idx]);
}

// ---------------- Kernel 0b: q^T (bf16), LDS-tiled transpose ----------------
__global__ void transpose_kernel(const unsigned short* __restrict__ qb,
                                 unsigned short* __restrict__ qT) {
  __shared__ unsigned short tile[32][33];
  const int t0 = blockIdx.x * 32, d0 = blockIdx.y * 32, b = blockIdx.z;
  const int tx = threadIdx.x, ty = threadIdx.y;
  for (int j = ty; j < 32; j += 8)
    tile[j][tx] = qb[((size_t)b * L_ + (t0 + j)) * D_ + d0 + tx];
  __syncthreads();
  for (int j = ty; j < 32; j += 8)
    qT[((size_t)b * D_ + (d0 + j)) * L_ + t0 + tx] = tile[tx][j];
}

// ---------------- Kernel 1: fused scores + row softmax ----------------
// One block (8 waves) per 16-row s-tile. A-operands (h rows + w2 rows) are staged
// once into LDS via async-to-LDS loads; each wave computes a 16x128 slab of the
// score row-block with bf16 WMMA; the full 16x1024 f32 score tile lives in LDS,
// softmax is done in-block, p (bf16, normalized) and row-max m go to global.
__global__ void __launch_bounds__(256)
scores_softmax_kernel(const unsigned short* __restrict__ hb,
                      const unsigned short* __restrict__ qkb,
                      const unsigned short* __restrict__ qb,
                      const unsigned short* __restrict__ w2c,
                      const float* __restrict__ b1,
                      const float* __restrict__ b2,
                      unsigned short* __restrict__ p_buf,
                      float* __restrict__ m_buf) {
  __shared__ unsigned short stageA[2 * 16 * D_];  // [0..] h rows, [16*D_..] w2 rows (48 KB)
  __shared__ float atile[16 * L_];                // 16x1024 f32 scores (64 KB)
  __shared__ float red[16][17];

  const int tid  = threadIdx.x;
  const int lane = tid & 31;
  const int wid  = tid >> 5;
  const int task = blockIdx.x;            // B * L/16 = 1024 blocks
  const int b    = task >> 6;
  const int s0   = (task & 63) << 4;
  const int half = lane >> 4;
  const int l16  = lane & 15;

  // ---- stage A-operands into LDS with async-to-LDS (TDM-class) loads ----
  {
    const unsigned short* hsrc = hb  + ((size_t)b * L_ + s0) * D_;
    const unsigned short* wsrc = w2c + (size_t)s0 * D_;
    const unsigned int lds_base = (unsigned int)(uintptr_t)(&stageA[0]);
    // 2 regions * 16 rows * 96 chunks of 16B = 3072 chunks; 12 uniform passes
#pragma unroll 1
    for (int c = tid; c < 3072; c += 256) {
      const int cc  = c & 1535;
      const unsigned short* g =
          ((c < 1536) ? (hsrc + (size_t)(cc / 96) * D_ + (cc % 96) * 8)
                      : (wsrc + (size_t)(cc / 96) * D_ + (cc % 96) * 8));
      const unsigned int ldsa = lds_base + (unsigned int)c * 16u;
      const unsigned long long ga = (unsigned long long)(uintptr_t)g;
      asm volatile("global_load_async_to_lds_b128 %0, %1, off"
                   :: "v"(ldsa), "v"(ga) : "memory");
    }
    asm volatile("s_wait_asynccnt 0" ::: "memory");
  }
  __syncthreads();

  // ---- GEMM: each wave computes 16 x 128 (8 t-tiles) of a = h@qk^T + w2@q^T ----
  const unsigned short* shH = &stageA[(size_t)l16 * D_ + half * 8];
  const unsigned short* shW = &stageA[(size_t)(16 + l16) * D_ + half * 8];
  float bs[8];
#pragma unroll
  for (int r2 = 0; r2 < 8; ++r2) bs[r2] = b2[s0 + r2 + half * 8];

  for (int tt = 0; tt < 8; ++tt) {
    const int t0 = wid * 128 + tt * 16;
    const unsigned short* qkrow = qkb + ((size_t)b * L_ + t0 + l16) * D_ + half * 8;
    const unsigned short* qrow  = qb  + ((size_t)b * L_ + t0 + l16) * D_ + half * 8;
    v8f acc = {};
#pragma unroll 4
    for (int k = 0; k < D_; k += 32) {       // h . qk^T  (covers w1h + w3hq)
      v16bf av = load_a_tile(shH + k);       // ds_load from LDS stage
      v16bf bv = load_b_tile(qkrow + k);
      acc = __builtin_amdgcn_wmma_f32_16x16x32_bf16(false, av, false, bv,
                                                    (short)0, acc, false, false);
    }
#pragma unroll 4
    for (int k = 0; k < D_; k += 32) {       // w2 . q^T  (the swapaxes(w2q) term)
      v16bf av = load_a_tile(shW + k);
      v16bf bv = load_b_tile(qrow + k);
      acc = __builtin_amdgcn_wmma_f32_16x16x32_bf16(false, av, false, bv,
                                                    (short)0, acc, false, false);
    }
    const float bt = b1[t0 + l16];
#pragma unroll
    for (int r2 = 0; r2 < 8; ++r2)           // C layout: M = r2 + 8*half, N = l16
      atile[(r2 + 8 * half) * L_ + t0 + l16] = acc[r2] + bt + bs[r2];
  }
  __syncthreads();

  // ---- in-block softmax over the 16 x 1024 tile ----
  const int row = tid >> 4;   // 16 threads per row
  const int sub = tid & 15;
  float* arow = &atile[row * L_];

  float lm = -3.402823466e+38f;
  for (int t = sub; t < L_; t += 16) lm = fmaxf(lm, arow[t]);
  red[row][sub] = lm;
  __syncthreads();
  if (sub == 0) {
    float m = red[row][0];
#pragma unroll
    for (int j = 1; j < 16; ++j) m = fmaxf(m, red[row][j]);
    red[row][16] = m;
    m_buf[(size_t)b * L_ + s0 + row] = m;    // m = max_t a (needed for p2)
  }
  __syncthreads();
  const float m = red[row][16];

  float ls = 0.f;
  for (int t = sub; t < L_; t += 16) {
    float e = __expf(arow[t] - m);
    arow[t] = e;
    ls += e;
  }
  __syncthreads();                            // red[row][16] consumed by all
  red[row][sub] = ls;
  __syncthreads();
  if (sub == 0) {
    float s = red[row][0];
#pragma unroll
    for (int j = 1; j < 16; ++j) s += red[row][j];
    red[row][16] = 1.0f / s;
  }
  __syncthreads();
  const float inv = red[row][16];

  unsigned short* prow = p_buf + ((size_t)b * L_ + s0 + row) * L_;
  for (int t = sub; t < L_; t += 16) prow[t] = f2bf(arow[t] * inv);
}

// ---------------- Kernel 2: p2 = softmax over s of m, per batch ----------------
__global__ void p2_kernel(const float* __restrict__ m_buf, float* __restrict__ p2) {
  const int b = blockIdx.x;
  const float* m = m_buf + (size_t)b * L_;
  __shared__ float red[256];
  float v[4];
#pragma unroll
  for (int i = 0; i < 4; ++i) v[i] = m[threadIdx.x + i * 256];
  red[threadIdx.x] = fmaxf(fmaxf(v[0], v[1]), fmaxf(v[2], v[3]));
  __syncthreads();
  for (int off = 128; off > 0; off >>= 1) {
    if (threadIdx.x < off) red[threadIdx.x] = fmaxf(red[threadIdx.x], red[threadIdx.x + off]);
    __syncthreads();
  }
  const float mx = red[0];
  __syncthreads();
  float ls = 0.f;
#pragma unroll
  for (int i = 0; i < 4; ++i) { v[i] = __expf(v[i] - mx); ls += v[i]; }
  red[threadIdx.x] = ls;
  __syncthreads();
  for (int off = 128; off > 0; off >>= 1) {
    if (threadIdx.x < off) red[threadIdx.x] += red[threadIdx.x + off];
    __syncthreads();
  }
  const float inv = 1.0f / red[0];
#pragma unroll
  for (int i = 0; i < 4; ++i)
    p2[(size_t)b * L_ + threadIdx.x + i * 256] = v[i] * inv;
}

// ---------------- Kernel 3: c = p @ q, fused output assembly ----------------
__global__ void out_kernel(const unsigned short* __restrict__ p_buf,
                           const unsigned short* __restrict__ qT,
                           const float* __restrict__ h,
                           const float* __restrict__ p2,
                           float* __restrict__ out) {
  const int lane = threadIdx.x & 31;
  const int wid  = threadIdx.x >> 5;
  const int task = blockIdx.x * 8 + wid;        // B * (L/16) * (D/16) = 49152 tasks
  const int b    = task / (64 * 48);
  const int rem  = task % (64 * 48);
  const int s0   = (rem / 48) << 4;
  const int d0   = (rem % 48) << 4;
  const int half = lane >> 4;
  const int l16  = lane & 15;

  const unsigned short* prow = p_buf + ((size_t)b * L_ + s0 + l16) * L_ + half * 8;
  const unsigned short* qrow = qT    + ((size_t)b * D_ + d0 + l16) * L_ + half * 8;
  v8f acc = {};
#pragma unroll 4
  for (int k = 0; k < L_; k += 32) {            // c = P @ Q  (B from q^T rows)
    v16bf av = load_a_tile(prow + k);
    v16bf bv = load_b_tile(qrow + k);
    acc = __builtin_amdgcn_wmma_f32_16x16x32_bf16(false, av, false, bv,
                                                  (short)0, acc, false, false);
  }
#pragma unroll
  for (int r2 = 0; r2 < 8; ++r2) {
    const int s = s0 + r2 + half * 8;
    const int d = d0 + l16;
    const size_t rowi = (size_t)b * L_ + s;
    const float hv = h[rowi * D_ + d];
    const float cv = acc[r2];
    const float sc = p2[rowi];
    float* o = out + rowi * (size_t)(4 * D_);
    o[d]          = hv;
    o[D_ + d]     = cv;
    o[2 * D_ + d] = hv * cv;
    o[3 * D_ + d] = hv * sc * cv;
  }
}

extern "C" void kernel_launch(void* const* d_in, const int* in_sizes, int n_in,
                              void* d_out, int out_size, void* d_ws, size_t ws_size,
                              hipStream_t stream) {
  const float* h   = (const float*)d_in[0];
  const float* q   = (const float*)d_in[1];
  const float* w1w = (const float*)d_in[2];
  const float* w1b = (const float*)d_in[3];
  const float* w2w = (const float*)d_in[4];
  const float* w2b = (const float*)d_in[5];
  float* out = (float*)d_out;

  const size_t N_BLD = (size_t)B_ * L_ * D_;      // 12,582,912
  const size_t N_LD  = (size_t)L_ * D_;           // 786,432
  const size_t N_BLL = (size_t)B_ * L_ * L_;      // 16,777,216

  uint8_t* ws = (uint8_t*)d_ws;
  size_t off = 0;
  unsigned short* hb  = (unsigned short*)(ws + off); off += N_BLD * 2;
  unsigned short* qkb = (unsigned short*)(ws + off); off += N_BLD * 2;
  unsigned short* qb  = (unsigned short*)(ws + off); off += N_BLD * 2;
  unsigned short* qT  = (unsigned short*)(ws + off); off += N_BLD * 2;
  unsigned short* w2c = (unsigned short*)(ws + off); off += N_LD * 2;
  unsigned short* p_buf = (unsigned short*)(ws + off); off += N_BLL * 2;
  float*          m_buf = (float*)(ws + off);        off += (size_t)B_ * L_ * 4;
  float*          p2b   = (float*)(ws + off);        off += (size_t)B_ * L_ * 4;

  // 0) bf16 conversions + qk = q + w1_w
  prep_kernel<<<(unsigned)((N_BLD + 255) / 256), 256, 0, stream>>>(
      h, q, w1w, w2w, hb, qb, qkb, w2c);
  // 0b) q^T
  transpose_kernel<<<dim3(L_ / 32, D_ / 32, B_), dim3(32, 8), 0, stream>>>(qb, qT);
  // 1) fused scores + softmax (WMMA + async-to-LDS staging)
  scores_softmax_kernel<<<B_ * (L_ / 16), 256, 0, stream>>>(
      hb, qkb, qb, w2c, w1b, w2b, p_buf, m_buf);
  // 2) p2 = softmax(m) over s
  p2_kernel<<<B_, 256, 0, stream>>>(m_buf, p2b);
  // 3) c = p@q (WMMA) + output assembly
  out_kernel<<<(B_ * (L_ / 16) * (D_ / 16)) / 8, 256, 0, stream>>>(
      p_buf, qT, h, p2b, out);
}
